// SwinTransformerBlock_69449621176983
// MI455X (gfx1250) — matrix operand order repaired and checked by reading, not compiled, not run on hardware
//
#include <hip/hip_runtime.h>
#include <hip/hip_bf16.h>
#include <math.h>

typedef _Float16 half_t;
typedef __attribute__((ext_vector_type(16))) _Float16 v16h;
typedef __attribute__((ext_vector_type(8)))  _Float16 v8h;
typedef __attribute__((ext_vector_type(8)))  float    v8f;

#define DEV static __device__ __forceinline__

DEV float wave_sum(float v) {
  #pragma unroll
  for (int m = 16; m >= 1; m >>= 1) v += __shfl_xor(v, m, 32);
  return v;
}

DEV int hreg(int c) { return c < 49 ? 0 : (c < 53 ? 1 : 2); }

DEV v16h make_frag(v8h lo, v8h hi) {
  v16h r;
  #pragma unroll
  for (int i = 0; i < 8; i++) { r[i] = lo[i]; r[8 + i] = hi[i]; }
  return r;
}

// ---------------------------------------------------------------------------
// Weight fp32 [K][N] -> f16 transposed [N][K]
// ---------------------------------------------------------------------------
__global__ __launch_bounds__(256) void cvt_t_kernel(const float* __restrict__ w,
                                                    half_t* __restrict__ wt,
                                                    int K, int N) {
  int idx = blockIdx.x * 256 + threadIdx.x;
  if (idx >= K * N) return;
  int k = idx / N, n = idx - k * N;
  wt[(size_t)n * K + k] = (half_t)w[idx];
}

// ---------------------------------------------------------------------------
// LN1 + cyclic shift(-3,-3) + window partition -> f16 [50176, 512]
// One wave per output token row.
// ---------------------------------------------------------------------------
__global__ __launch_bounds__(256) void ln1_kernel(const float* __restrict__ x,
                                                  const float* __restrict__ gam,
                                                  const float* __restrict__ bet,
                                                  half_t* __restrict__ out) {
  int wave = threadIdx.x >> 5, L = threadIdx.x & 31;
  int t = blockIdx.x * 8 + wave;               // token id 0..50175
  int b = t / 3136, rem = t - b * 3136;
  int win = rem / 49, loc = rem - win * 49;
  int wh = win >> 3, ww = win & 7;
  int ii = loc / 7, jj = loc - ii * 7;
  int hs = (wh * 7 + ii + 3) % 56;             // roll(-3): h[a] = x[(a+3)%56]
  int ws = (ww * 7 + jj + 3) % 56;
  const float* row = x + ((size_t)b * 3136 + hs * 56 + ws) * 512;
  float vals[16], s = 0.f, ss = 0.f;
  #pragma unroll
  for (int e = 0; e < 16; e++) {
    float v = row[e * 32 + L];
    vals[e] = v; s += v; ss += v * v;
  }
  s = wave_sum(s); ss = wave_sum(ss);
  float mean = s * (1.f / 512.f);
  float rstd = rsqrtf(ss * (1.f / 512.f) - mean * mean + 1e-5f);
  half_t* orow = out + (size_t)t * 512;
  #pragma unroll
  for (int e = 0; e < 16; e++) {
    int c = e * 32 + L;
    orow[c] = (half_t)((vals[e] - mean) * rstd * gam[c] + bet[c]);
  }
}

// ---------------------------------------------------------------------------
// window reverse + roll(+3,+3) + residual add; then LN2 -> f16
// One wave per spatial row (b,ho,wo).
// ---------------------------------------------------------------------------
__global__ __launch_bounds__(256) void res_ln2_kernel(const float* __restrict__ x,
                                                      const float* __restrict__ proj_out,
                                                      const float* __restrict__ gam,
                                                      const float* __restrict__ bet,
                                                      float* __restrict__ x1,
                                                      half_t* __restrict__ h2) {
  int wave = threadIdx.x >> 5, L = threadIdx.x & 31;
  int idx = blockIdx.x * 8 + wave;             // spatial row id 0..50175
  int b = idx / 3136, rem = idx - b * 3136;
  int ho = rem / 56, wo = rem - ho * 56;
  int hs = (ho + 53) % 56, ws = (wo + 53) % 56;     // reverse roll(+3)
  int t = (b * 64 + (hs / 7) * 8 + (ws / 7)) * 49 + (hs % 7) * 7 + (ws % 7);
  const float* xr = x + (size_t)idx * 512;
  const float* pr = proj_out + (size_t)t * 512;
  float vals[16], s = 0.f, ss = 0.f;
  #pragma unroll
  for (int e = 0; e < 16; e++) {
    int c = e * 32 + L;
    float v = xr[c] + pr[c];
    vals[e] = v; s += v; ss += v * v;
  }
  float* x1r = x1 + (size_t)idx * 512;
  #pragma unroll
  for (int e = 0; e < 16; e++) x1r[e * 32 + L] = vals[e];
  s = wave_sum(s); ss = wave_sum(ss);
  float mean = s * (1.f / 512.f);
  float rstd = rsqrtf(ss * (1.f / 512.f) - mean * mean + 1e-5f);
  half_t* hr = h2 + (size_t)idx * 512;
  #pragma unroll
  for (int e = 0; e < 16; e++) {
    int c = e * 32 + L;
    hr[c] = (half_t)((vals[e] - mean) * rstd * gam[c] + bet[c]);
  }
}

// ---------------------------------------------------------------------------
// WMMA GEMM: C[M,N] = A[M,K](f16) * Bt[N,K](f16)^T + bias, optional epilogues.
// Block = 128 threads (4 waves), 128x64 C tile; each wave: 2 A-frag x 4 B-frag
// = 8 WMMAs per K-step with all fragments preloaded before the WMMA chain.
// mode: 0 = f16 out, 1 = f16 out + exact GELU, 2 = f32 out, 3 = f32 + res add
// ---------------------------------------------------------------------------
__global__ __launch_bounds__(128) void gemm_kernel(const half_t* __restrict__ A,
                                                   const half_t* __restrict__ Bt,
                                                   const float* __restrict__ bias,
                                                   const float* __restrict__ res,
                                                   void* __restrict__ Cout,
                                                   int M, int N, int K, int mode) {
  __shared__ __align__(16) half_t As[128 * 40];
  __shared__ __align__(16) half_t Bs[64 * 40];
  int tid  = threadIdx.x;
  int wave = tid >> 5, L = tid & 31;
  int g = (L >> 4) & 1, lm = L & 15;
  int n0 = blockIdx.x * 64, m0 = blockIdx.y * 128;

  v8f acc[2][4] = {};

  for (int k0 = 0; k0 < K; k0 += 32) {
    v8h a_frag[4], b_frag[2];
    #pragma unroll
    for (int it = 0; it < 4; it++) {
      int c = tid + it * 128;
      int row = c >> 2, cc = (c & 3) * 8;
      a_frag[it] = *(const v8h*)(A + (size_t)(m0 + row) * K + k0 + cc);
    }
    #pragma unroll
    for (int it = 0; it < 2; it++) {
      int c = tid + it * 128;
      int row = c >> 2, cc = (c & 3) * 8;
      b_frag[it] = *(const v8h*)(Bt + (size_t)(n0 + row) * K + k0 + cc);
    }
    __syncthreads();
    #pragma unroll
    for (int it = 0; it < 4; it++) {
      int c = tid + it * 128;
      int row = c >> 2, cc = (c & 3) * 8;
      *(v8h*)(As + row * 40 + cc) = a_frag[it];
    }
    #pragma unroll
    for (int it = 0; it < 2; it++) {
      int c = tid + it * 128;
      int row = c >> 2, cc = (c & 3) * 8;
      *(v8h*)(Bs + row * 40 + cc) = b_frag[it];
    }
    __syncthreads();

    // preload ALL fragments, then issue the 8 WMMAs back-to-back
    v16h av[2], bv[4];
    #pragma unroll
    for (int mi = 0; mi < 2; mi++) {
      int m = wave * 32 + mi * 16 + lm;
      av[mi] = make_frag(*(const v8h*)(As + m * 40 + 8 * g),
                         *(const v8h*)(As + m * 40 + 16 + 8 * g));
    }
    #pragma unroll
    for (int nt = 0; nt < 4; nt++) {
      int n = nt * 16 + lm;
      bv[nt] = make_frag(*(const v8h*)(Bs + n * 40 + 8 * g),
                         *(const v8h*)(Bs + n * 40 + 16 + 8 * g));
    }
    #pragma unroll
    for (int mi = 0; mi < 2; mi++)
      #pragma unroll
      for (int nt = 0; nt < 4; nt++)
        acc[mi][nt] = __builtin_amdgcn_wmma_f32_16x16x32_f16(
            false, av[mi], false, bv[nt], (short)0, acc[mi][nt], false, false);
  }

  #pragma unroll
  for (int mi = 0; mi < 2; mi++) {
    #pragma unroll
    for (int nt = 0; nt < 4; nt++) {
      int n = n0 + nt * 16 + lm;
      float bv = bias[n];
      #pragma unroll
      for (int r = 0; r < 8; r++) {
        int m = m0 + wave * 32 + mi * 16 + 8 * g + r;
        float v = acc[mi][nt][r] + bv;
        if (mode == 1) v = 0.5f * v * (1.f + erff(v * 0.70710678118f));
        size_t off = (size_t)m * N + n;
        if (mode >= 2) {
          float o = v;
          if (mode == 3) o += res[off];
          ((float*)Cout)[off] = o;
        } else {
          ((half_t*)Cout)[off] = (half_t)v;
        }
      }
    }
  }
}

// ---------------------------------------------------------------------------
// Windowed attention, one wave per (window, head).
// S = (Q*scale)K^T + rpb + mask ; softmax ; O = P V   (all matmuls via WMMA)
// B-fragments are loop-invariant and hoisted before the WMMA chains.
// ---------------------------------------------------------------------------
__global__ __launch_bounds__(32) void attn_kernel(const half_t* __restrict__ qkv,
                                                  const float* __restrict__ rpb,
                                                  half_t* __restrict__ o) {
  __shared__ __align__(16) half_t q_s[64 * 32];
  __shared__ __align__(16) half_t k_s[64 * 32];
  __shared__ __align__(16) half_t vT_s[32 * 64];
  __shared__ __align__(16) float  s_s[64 * 64];
  __shared__ __align__(16) half_t p_s[64 * 64];

  int L = threadIdx.x;
  int w = blockIdx.x >> 4, head = blockIdx.x & 15;
  int g = L >> 4, lm = L & 15;

  for (int t = L; t < 64 * 32; t += 32) { q_s[t] = (half_t)0; k_s[t] = (half_t)0; }
  for (int t = L; t < 32 * 64; t += 32) vT_s[t] = (half_t)0;
  __syncthreads();

  // load q/k rows, v transposed
  for (int t = L; t < 49 * 4; t += 32) {
    int row = t >> 2, ch = (t & 3) * 8;
    size_t base = ((size_t)w * 49 + row) * 1536 + head * 32 + ch;
    *(v8h*)(q_s + row * 32 + ch) = *(const v8h*)(qkv + base);
    *(v8h*)(k_s + row * 32 + ch) = *(const v8h*)(qkv + base + 512);
    v8h vv = *(const v8h*)(qkv + base + 1024);
    #pragma unroll
    for (int e = 0; e < 8; e++) vT_s[(ch + e) * 64 + row] = vv[e];
  }
  __syncthreads();

  int win = w & 63, wh = win >> 3, ww = win & 7;
  const float scale = 0.17677669529663687f;   // 32^-0.5

  // ---- S = Q K^T ----
  v16h kb[4];
  #pragma unroll
  for (int nj = 0; nj < 4; nj++) {
    int j = nj * 16 + lm;
    kb[nj] = make_frag(*(const v8h*)(k_s + j * 32 + 8 * g),
                       *(const v8h*)(k_s + j * 32 + 16 + 8 * g));
  }
  for (int mi = 0; mi < 4; mi++) {
    int m = mi * 16 + lm;
    v16h av = make_frag(*(const v8h*)(q_s + m * 32 + 8 * g),
                        *(const v8h*)(q_s + m * 32 + 16 + 8 * g));
    v8f sacc[4];
    #pragma unroll
    for (int nj = 0; nj < 4; nj++) {
      v8f z = {};
      sacc[nj] = __builtin_amdgcn_wmma_f32_16x16x32_f16(
          false, av, false, kb[nj], (short)0, z, false, false);
    }
    #pragma unroll
    for (int nj = 0; nj < 4; nj++) {
      int j = nj * 16 + lm;
      #pragma unroll
      for (int r = 0; r < 8; r++) {
        int i = mi * 16 + 8 * g + r;
        float sv = -1e30f;
        if (i < 49 && j < 49) {
          int yi = i / 7, xi = i - yi * 7;
          int yj = j / 7, xj = j - yj * 7;
          // faithful to reference: idx = (dx+6)*13 + (dy+6)
          int ridx = (xi - xj + 6) * 13 + (yi - yj + 6);
          float rb = rpb[ridx * 16 + head];
          int gi = hreg(wh * 7 + yi) * 3 + hreg(ww * 7 + xi);
          int gj = hreg(wh * 7 + yj) * 3 + hreg(ww * 7 + xj);
          sv = sacc[nj][r] * scale + rb + ((gi != gj) ? -100.f : 0.f);
        }
        s_s[i * 64 + j] = sv;
      }
    }
  }
  __syncthreads();

  // ---- softmax over j (rows split across lanes) ----
  for (int row = L; row < 49; row += 32) {
    float mx = -1e30f;
    for (int j = 0; j < 49; j++) mx = fmaxf(mx, s_s[row * 64 + j]);
    float sum = 0.f;
    for (int j = 0; j < 49; j++) sum += __expf(s_s[row * 64 + j] - mx);
    float inv = 1.f / sum;
    for (int j = 0; j < 49; j++)
      p_s[row * 64 + j] = (half_t)(__expf(s_s[row * 64 + j] - mx) * inv);
    for (int j = 49; j < 64; j++) p_s[row * 64 + j] = (half_t)0;
  }
  for (int t = L; t < 15 * 64; t += 32) p_s[49 * 64 + t] = (half_t)0;
  __syncthreads();

  // ---- O = P V ----
  v16h vb[2][2];
  #pragma unroll
  for (int kk = 0; kk < 2; kk++)
    #pragma unroll
    for (int nt = 0; nt < 2; nt++) {
      int n = nt * 16 + lm;
      vb[kk][nt] = make_frag(*(const v8h*)(vT_s + n * 64 + kk * 32 + 8 * g),
                             *(const v8h*)(vT_s + n * 64 + kk * 32 + 16 + 8 * g));
    }
  for (int mi = 0; mi < 4; mi++) {
    int m = mi * 16 + lm;
    v16h av[2];
    #pragma unroll
    for (int kk = 0; kk < 2; kk++)
      av[kk] = make_frag(*(const v8h*)(p_s + m * 64 + kk * 32 + 8 * g),
                         *(const v8h*)(p_s + m * 64 + kk * 32 + 16 + 8 * g));
    v8f acc[2] = {};
    #pragma unroll
    for (int kk = 0; kk < 2; kk++)
      #pragma unroll
      for (int nt = 0; nt < 2; nt++)
        acc[nt] = __builtin_amdgcn_wmma_f32_16x16x32_f16(
            false, av[kk], false, vb[kk][nt], (short)0, acc[nt], false, false);
    #pragma unroll
    for (int nt = 0; nt < 2; nt++) {
      int d = nt * 16 + lm;
      #pragma unroll
      for (int r = 0; r < 8; r++) {
        int i = mi * 16 + 8 * g + r;
        if (i < 49)
          o[((size_t)w * 49 + i) * 512 + head * 32 + d] = (half_t)acc[nt][r];
      }
    }
  }
}

// ---------------------------------------------------------------------------
extern "C" void kernel_launch(void* const* d_in, const int* in_sizes, int n_in,
                              void* d_out, int out_size, void* d_ws, size_t ws_size,
                              hipStream_t stream) {
  (void)in_sizes; (void)n_in; (void)out_size; (void)ws_size;
  const float* x      = (const float*)d_in[0];
  const float* ln1_s  = (const float*)d_in[1];
  const float* ln1_b  = (const float*)d_in[2];
  const float* qkv_w  = (const float*)d_in[3];
  const float* qkv_b  = (const float*)d_in[4];
  const float* rpb    = (const float*)d_in[5];
  const float* proj_w = (const float*)d_in[6];
  const float* proj_b = (const float*)d_in[7];
  const float* ln2_s  = (const float*)d_in[8];
  const float* ln2_b  = (const float*)d_in[9];
  const float* fc1_w  = (const float*)d_in[10];
  const float* fc1_b  = (const float*)d_in[11];
  const float* fc2_w  = (const float*)d_in[12];
  const float* fc2_b  = (const float*)d_in[13];
  float* out = (float*)d_out;

  const size_t T = 50176;
  char* p = (char*)d_ws;
  auto alloc = [&](size_t bytes) -> char* {
    char* r = p; p += (bytes + 255) & ~(size_t)255; return r;
  };
  half_t* h16   = (half_t*)alloc(T * 512 * sizeof(half_t));
  half_t* qkv16 = (half_t*)alloc(T * 1536 * sizeof(half_t));
  half_t* o16   = (half_t*)alloc(T * 512 * sizeof(half_t));
  float*  projo = (float* )alloc(T * 512 * sizeof(float));
  float*  x1    = (float* )alloc(T * 512 * sizeof(float));
  half_t* h2    = (half_t*)alloc(T * 512 * sizeof(half_t));
  half_t* act16 = (half_t*)alloc(T * 2048 * sizeof(half_t));
  half_t* qkvT  = (half_t*)alloc(512 * 1536 * sizeof(half_t));
  half_t* projT = (half_t*)alloc(512 * 512 * sizeof(half_t));
  half_t* fc1T  = (half_t*)alloc(512 * 2048 * sizeof(half_t));
  half_t* fc2T  = (half_t*)alloc(2048 * 512 * sizeof(half_t));

  // weight conversion (transposed f16)
  cvt_t_kernel<<<(512 * 1536 + 255) / 256, 256, 0, stream>>>(qkv_w, qkvT, 512, 1536);
  cvt_t_kernel<<<(512 * 512  + 255) / 256, 256, 0, stream>>>(proj_w, projT, 512, 512);
  cvt_t_kernel<<<(512 * 2048 + 255) / 256, 256, 0, stream>>>(fc1_w, fc1T, 512, 2048);
  cvt_t_kernel<<<(2048 * 512 + 255) / 256, 256, 0, stream>>>(fc2_w, fc2T, 2048, 512);

  // LN1 + shift + partition
  ln1_kernel<<<6272, 256, 0, stream>>>(x, ln1_s, ln1_b, h16);

  // qkv GEMM
  gemm_kernel<<<dim3(24, 392), 128, 0, stream>>>(h16, qkvT, qkv_b, nullptr,
                                                 qkv16, 50176, 1536, 512, 0);
  // windowed attention
  attn_kernel<<<16384, 32, 0, stream>>>(qkv16, rpb, o16);

  // proj GEMM (f32 out, token order)
  gemm_kernel<<<dim3(8, 392), 128, 0, stream>>>(o16, projT, proj_b, nullptr,
                                                projo, 50176, 512, 512, 2);
  // reverse + residual + LN2
  res_ln2_kernel<<<6272, 256, 0, stream>>>(x, projo, ln2_s, ln2_b, x1, h2);

  // fc1 GEMM + GELU
  gemm_kernel<<<dim3(32, 392), 128, 0, stream>>>(h2, fc1T, fc1_b, nullptr,
                                                 act16, 50176, 2048, 512, 1);
  // fc2 GEMM + residual -> out
  gemm_kernel<<<dim3(8, 392), 128, 0, stream>>>(act16, fc2T, fc2_b, x1,
                                                out, 50176, 512, 2048, 3);
}
